// SimpleEncoder_22127671509186
// MI455X (gfx1250) — matrix operand order repaired and checked by reading, not compiled, not run on hardware
//
#include <hip/hip_runtime.h>

// ---------------------------------------------------------------------------
// Point-group encoder for MI455X (gfx1250, wave32, WMMA).
// Pipeline: FPS -> KNN -> gather+L1(+BN1 stats) -> BN1 finalize ->
//           L2 + group-max + concat + L3 (+BN2 stats, fused) -> BN2 finalize ->
//           L4 + group-max -> features.
// GEMMs use v_wmma_f32_16x16x32_f16 (f32 accumulate). Intermediates stored f16.
// ---------------------------------------------------------------------------

#define BB   16      // batches
#define NN   8192    // points per batch
#define GG   512     // groups per batch
#define KNB  32      // neighbors per group (= rows per group)
#define MG   (BB*GG) // total groups = 8192
#define C1_  128
#define C2_  256
#define C3_  512
#define C4_  128

typedef __attribute__((ext_vector_type(16))) _Float16 v16h;
typedef __attribute__((ext_vector_type(8)))  _Float16 v8h;
typedef __attribute__((ext_vector_type(8)))  float    v8f;

// ---- CDNA5 WMMA operand loaders (contiguous k order, same in A and B) ----------------------

// A matrix 16x32 f16 (MxK): lane<16 -> K base 0, lane>=16 -> K base 16;
// vector elems [0..15] = K {base..base+15} (contiguous, matches load_B16).
__device__ __forceinline__ v16h load_A16(const _Float16* base, int row0, int k0, int ld) {
  const int lane = threadIdx.x & 31;
  const int m    = lane & 15;
  const int kb   = (lane >> 4) << 4;          // 0 or 16
  const _Float16* p = base + (size_t)(row0 + m) * ld + k0 + kb;
  v8h lo = *(const v8h*)(p);
  v8h hi = *(const v8h*)(p + 8);
  v16h v;
#pragma unroll
  for (int j = 0; j < 8; ++j) { v[j] = lo[j]; v[j + 8] = hi[j]; }
  return v;
}

// B matrix 32x16 f16 (KxN), B[k][n] = W[n][k] (W row-major (out,in)):
// lane<16 -> K 0..15, lane>=16 -> K 16..31; 16 contiguous halfs per lane.
__device__ __forceinline__ v16h load_B16(const _Float16* W, int n0, int k0, int ld) {
  const int lane = threadIdx.x & 31;
  const int n    = lane & 15;
  const int kb   = (lane >> 4) << 4;          // 0 or 16
  const _Float16* p = W + (size_t)(n0 + n) * ld + k0 + kb;
  return *(const v16h*)(p);
}

__device__ __forceinline__ v8f wmma16(v16h a, v16h b, v8f c) {
  return __builtin_amdgcn_wmma_f32_16x16x32_f16(false, a, false, b, (short)0, c, false, false);
}

// ---------------------------------------------------------------------------
// FPS: one workgroup per batch; LDS running-min distance; 512 serial argmax.
// Emits index BEFORE the update (matches the lax.scan carry/output ordering).
// ---------------------------------------------------------------------------
__global__ void __launch_bounds__(256)
fps_kernel(const float* __restrict__ xyz, int* __restrict__ cidx,
           float* __restrict__ outCenter) {
  const int b = blockIdx.x, tid = threadIdx.x;
  __shared__ float dist[NN];
  __shared__ float redD[256];
  __shared__ int   redI[256];
  const float* X = xyz + (size_t)b * NN * 3;
  for (int i = tid; i < NN; i += 256) dist[i] = 1e10f;
  int far = 0;
  __syncthreads();
  for (int it = 0; it < GG; ++it) {
    if (tid == 0) cidx[b * GG + it] = far;
    if (tid < 3)  outCenter[((size_t)b * GG + it) * 3 + tid] = X[far * 3 + tid];
    const float cx = X[far * 3 + 0], cy = X[far * 3 + 1], cz = X[far * 3 + 2];
    float bd = -1.0f; int bi = 0;
    for (int i = tid; i < NN; i += 256) {
      const float dx = X[i * 3 + 0] - cx;
      const float dy = X[i * 3 + 1] - cy;
      const float dz = X[i * 3 + 2] - cz;
      const float dm = fminf(dist[i], dx * dx + dy * dy + dz * dz);
      dist[i] = dm;
      if (dm > bd) { bd = dm; bi = i; }
    }
    redD[tid] = bd; redI[tid] = bi;
    __syncthreads();
    for (int s = 128; s > 0; s >>= 1) {
      if (tid < s) {
        const float od = redD[tid + s]; const int oi = redI[tid + s];
        if (od > redD[tid] || (od == redD[tid] && oi < redI[tid])) { redD[tid] = od; redI[tid] = oi; }
      }
      __syncthreads();
    }
    far = redI[0];
    __syncthreads();
  }
}

// ---------------------------------------------------------------------------
// KNN: one workgroup per (b,g); distances in LDS; 32 masked-argmin rounds
// (lowest-index tie-break, matching lax.top_k ordering on -d).
// ---------------------------------------------------------------------------
__global__ void __launch_bounds__(256)
knn_kernel(const float* __restrict__ xyz, const float* __restrict__ centers,
           int* __restrict__ knnidx) {
  const int bg = blockIdx.x, b = bg >> 9, tid = threadIdx.x;
  __shared__ float dist[NN];
  __shared__ float redD[256];
  __shared__ int   redI[256];
  const float* X = xyz + (size_t)b * NN * 3;
  const float cx = centers[bg * 3 + 0], cy = centers[bg * 3 + 1], cz = centers[bg * 3 + 2];
  const float cn = cx * cx + cy * cy + cz * cz;
  for (int i = tid; i < NN; i += 256) {
    const float px = X[i * 3 + 0], py = X[i * 3 + 1], pz = X[i * 3 + 2];
    dist[i] = cn + (px * px + py * py + pz * pz) - 2.0f * (cx * px + cy * py + cz * pz);
  }
  __syncthreads();
  for (int kk = 0; kk < KNB; ++kk) {
    float bd = 3.4e38f; int bi = NN;
    for (int i = tid; i < NN; i += 256) {
      const float d = dist[i];
      if (d < bd) { bd = d; bi = i; }
    }
    redD[tid] = bd; redI[tid] = bi;
    __syncthreads();
    for (int s = 128; s > 0; s >>= 1) {
      if (tid < s) {
        const float od = redD[tid + s]; const int oi = redI[tid + s];
        if (od < redD[tid] || (od == redD[tid] && oi < redI[tid])) { redD[tid] = od; redI[tid] = oi; }
      }
      __syncthreads();
    }
    if (tid == 0) { knnidx[(size_t)bg * KNB + kk] = redI[0]; dist[redI[0]] = 3.4e38f; }
    __syncthreads();
  }
}

// ---------------------------------------------------------------------------
// Gather + L1 (3->128) + per-block BN1 partial stats (deterministic).
// K=3 is too small for WMMA; plain FMA. f1 stored f16 (stats from f32).
// ---------------------------------------------------------------------------
__global__ void __launch_bounds__(128)
gather_l1_kernel(const float* __restrict__ xyz, const int* __restrict__ knnidx,
                 const float* __restrict__ W1, const float* __restrict__ b1,
                 _Float16* __restrict__ f1h,
                 float* __restrict__ partS, float* __restrict__ partQ) {
  const int bg = blockIdx.x, b = bg >> 9, tid = threadIdx.x;
  __shared__ float pts[KNB * 3];
  if (tid < KNB * 3) {
    const int r = tid / 3, d = tid % 3;
    const int idx = knnidx[(size_t)bg * KNB + r];
    pts[tid] = xyz[((size_t)b * NN + idx) * 3 + d];
  }
  __syncthreads();
  const int o = tid;  // output channel 0..127
  const float w0 = W1[o * 3 + 0], w1 = W1[o * 3 + 1], w2 = W1[o * 3 + 2], bb = b1[o];
  _Float16* f1g = f1h + (size_t)bg * KNB * C1_;
  float s = 0.f, q = 0.f;
  for (int r = 0; r < KNB; ++r) {
    const float f = pts[r * 3 + 0] * w0 + pts[r * 3 + 1] * w1 + pts[r * 3 + 2] * w2 + bb;
    f1g[r * C1_ + o] = (_Float16)f;
    s += f; q += f * f;
  }
  partS[(size_t)bg * C1_ + o] = s;
  partQ[(size_t)bg * C1_ + o] = q;
}

// ---- deterministic 2-stage BN stat reduction + finalize -------------------
__global__ void bn_reduce_kernel(const float* __restrict__ partS, const float* __restrict__ partQ,
                                 float* __restrict__ midS, float* __restrict__ midQ,
                                 int C, int perChunk) {
  const int t = blockIdx.x * blockDim.x + threadIdx.x;
  if (t >= C * 32) return;
  const int c = t % C, chunk = t / C, b0 = chunk * perChunk;
  float s = 0.f, q = 0.f;
  for (int i = 0; i < perChunk; ++i) {
    s += partS[(size_t)(b0 + i) * C + c];
    q += partQ[(size_t)(b0 + i) * C + c];
  }
  midS[t] = s; midQ[t] = q;
}

__global__ void bn_finalize_kernel(const float* __restrict__ midS, const float* __restrict__ midQ,
                                   const float* __restrict__ gamma, const float* __restrict__ beta,
                                   float* __restrict__ scale, float* __restrict__ shift,
                                   int C, float invCnt) {
  const int c = blockIdx.x * blockDim.x + threadIdx.x;
  if (c >= C) return;
  float s = 0.f, q = 0.f;
  for (int ch = 0; ch < 32; ++ch) { s += midS[ch * C + c]; q += midQ[ch * C + c]; }
  const float mean = s * invCnt;
  const float var  = q * invCnt - mean * mean;
  const float sc   = gamma[c] * rsqrtf(var + 1e-5f);
  scale[c] = sc; shift[c] = beta[c] - mean * sc;
}

__global__ void f32_to_f16_kernel(const float* __restrict__ src, _Float16* __restrict__ dst, int n) {
  const int i = blockIdx.x * blockDim.x + threadIdx.x;
  if (i < n) dst[i] = (_Float16)src[i];
}

// ---------------------------------------------------------------------------
// Fused middle: BN1+ReLU -> L2 (128->256, WMMA) -> group max -> concat ->
// L3 (512->512, WMMA) -> f3 (f16) + deterministic BN2 partial stats.
// One workgroup (8 wave32s) per group of 32 points.
// ---------------------------------------------------------------------------
__global__ void __launch_bounds__(256)
encoder_mid_kernel(const _Float16* __restrict__ f1h,
                   const _Float16* __restrict__ W2h, const float* __restrict__ b2,
                   const _Float16* __restrict__ W3h, const float* __restrict__ b3,
                   const float* __restrict__ s1, const float* __restrict__ h1,
                   _Float16* __restrict__ f3h,
                   float* __restrict__ partS, float* __restrict__ partQ) {
  const int g = blockIdx.x, tid = threadIdx.x, w = tid >> 5, lane = tid & 31;
  __shared__ __align__(32) _Float16 A1[KNB * C1_];   //  8 KB: relu(bn1(f1))
  __shared__ __align__(32) _Float16 A2[KNB * C3_];   // 32 KB: [fg | f2] f16
  __shared__ float pS[4 * C3_];                      //  8 KB: per-slot stat partials
  __shared__ float pQ[4 * C3_];                      //  8 KB
  // Stage A1 = relu(bn1(f1_group)) as f16.
  const _Float16* f1g = f1h + (size_t)g * KNB * C1_;
  for (int e = tid; e < KNB * C1_; e += 256) {
    const int c = e & (C1_ - 1);
    float v = (float)f1g[e];
    v = fmaxf(v * s1[c] + h1[c], 0.f);
    A1[e] = (_Float16)v;
  }
  __syncthreads();
  // GEMM1: f2(32x256) = A1 @ W2^T + b2 -> A2 columns [256..512).
  for (int t = w * 4; t < w * 4 + 4; ++t) {           // 32 tiles / 8 waves
    const int mt = t >> 4, nt = t & 15;
    const int col = nt * 16 + (lane & 15);
    const float bias = b2[col];
    v8f acc;
#pragma unroll
    for (int j = 0; j < 8; ++j) acc[j] = bias;
#pragma unroll
    for (int kt = 0; kt < 4; ++kt) {
      v16h a  = load_A16(A1,  mt * 16, kt * 32, C1_);
      v16h bm = load_B16(W2h, nt * 16, kt * 32, C1_);
      acc = wmma16(a, bm, acc);
    }
    const int rbase = mt * 16 + ((lane >> 4) << 3);
#pragma unroll
    for (int j = 0; j < 8; ++j)
      A2[(rbase + j) * C3_ + C2_ + col] = (_Float16)acc[j];
  }
  __syncthreads();
  // fg = column max of f2; replicate into A2 columns [0..256) (concat order).
  if (tid < C2_) {
    float m = -3.4e38f;
    for (int r = 0; r < KNB; ++r) m = fmaxf(m, (float)A2[r * C3_ + C2_ + tid]);
    const _Float16 hm = (_Float16)m;
    for (int r = 0; r < KNB; ++r) A2[r * C3_ + tid] = hm;
  }
  __syncthreads();
  // GEMM2: f3(32x512) = A2 @ W3^T + b3 -> global f16 + stat partials.
  _Float16* f3g = f3h + (size_t)g * KNB * C3_;
  for (int t = w * 8; t < w * 8 + 8; ++t) {           // 64 tiles / 8 waves
    const int mt = t >> 5, nt = t & 31;
    const int col = nt * 16 + (lane & 15);
    const float bias = b3[col];
    v8f acc;
#pragma unroll
    for (int j = 0; j < 8; ++j) acc[j] = bias;
    for (int kt = 0; kt < 16; ++kt) {
      // Prefetch next W3 K-slice for this output row (gfx1250 global_prefetch_b8).
      __builtin_prefetch(W3h + (size_t)(nt * 16 + (lane & 15)) * C3_ + (((kt + 1) & 15) * 32), 0, 1);
      v16h a  = load_A16(A2,  mt * 16, kt * 32, C3_);
      v16h bm = load_B16(W3h, nt * 16, kt * 32, C3_);
      acc = wmma16(a, bm, acc);
    }
    const int rbase = mt * 16 + ((lane >> 4) << 3);
    float s = 0.f, q = 0.f;
#pragma unroll
    for (int j = 0; j < 8; ++j) {
      const float v = acc[j];
      f3g[(rbase + j) * C3_ + col] = (_Float16)v;
      s += v; q += v * v;
    }
    const int slot = mt * 2 + (lane >> 4);            // each (slot,col) written once
    pS[slot * C3_ + col] = s;
    pQ[slot * C3_ + col] = q;
  }
  __syncthreads();
  for (int c = tid; c < C3_; c += 256) {
    partS[(size_t)g * C3_ + c] = pS[c] + pS[C3_ + c] + pS[2 * C3_ + c] + pS[3 * C3_ + c];
    partQ[(size_t)g * C3_ + c] = pQ[c] + pQ[C3_ + c] + pQ[2 * C3_ + c] + pQ[3 * C3_ + c];
  }
}

// ---------------------------------------------------------------------------
// Final: BN2+ReLU -> L4 (512->128, WMMA) -> group max -> features.
// ---------------------------------------------------------------------------
__global__ void __launch_bounds__(256)
encoder_out_kernel(const _Float16* __restrict__ f3h,
                   const _Float16* __restrict__ W4h, const float* __restrict__ b4,
                   const float* __restrict__ s2, const float* __restrict__ h2,
                   float* __restrict__ out) {
  const int g = blockIdx.x, tid = threadIdx.x, w = tid >> 5, lane = tid & 31;
  __shared__ __align__(32) _Float16 A[KNB * C3_];    // 32 KB
  __shared__ float f4[KNB * C4_];                    // 16 KB
  const _Float16* f3g = f3h + (size_t)g * KNB * C3_;
  for (int e = tid; e < KNB * C3_; e += 256) {
    const int c = e & (C3_ - 1);
    float v = (float)f3g[e];
    v = fmaxf(v * s2[c] + h2[c], 0.f);
    A[e] = (_Float16)v;
  }
  __syncthreads();
  for (int t = w * 2; t < w * 2 + 2; ++t) {           // 16 tiles / 8 waves
    const int mt = t >> 3, nt = t & 7;
    const int col = nt * 16 + (lane & 15);
    const float bias = b4[col];
    v8f acc;
#pragma unroll
    for (int j = 0; j < 8; ++j) acc[j] = bias;
    for (int kt = 0; kt < 16; ++kt) {
      v16h a  = load_A16(A,   mt * 16, kt * 32, C3_);
      v16h bm = load_B16(W4h, nt * 16, kt * 32, C3_);
      acc = wmma16(a, bm, acc);
    }
    const int rbase = mt * 16 + ((lane >> 4) << 3);
#pragma unroll
    for (int j = 0; j < 8; ++j)
      f4[(rbase + j) * C4_ + col] = acc[j];
  }
  __syncthreads();
  if (tid < C4_) {
    float m = -3.4e38f;
    for (int r = 0; r < KNB; ++r) m = fmaxf(m, f4[r * C4_ + tid]);
    out[(size_t)BB * GG * 3 + (size_t)g * C4_ + tid] = m;   // features after centers
  }
}

// ---------------------------------------------------------------------------
extern "C" void kernel_launch(void* const* d_in, const int* in_sizes, int n_in,
                              void* d_out, int out_size, void* d_ws, size_t ws_size,
                              hipStream_t stream) {
  (void)in_sizes; (void)n_in; (void)out_size; (void)ws_size;
  const float* xyz = (const float*)d_in[0];
  const float* W1  = (const float*)d_in[1];
  const float* b1  = (const float*)d_in[2];
  const float* g1  = (const float*)d_in[3];
  const float* be1 = (const float*)d_in[4];
  const float* W2  = (const float*)d_in[5];
  const float* b2  = (const float*)d_in[6];
  const float* W3  = (const float*)d_in[7];
  const float* b3  = (const float*)d_in[8];
  const float* g2  = (const float*)d_in[9];
  const float* be2 = (const float*)d_in[10];
  const float* W4  = (const float*)d_in[11];
  const float* b4  = (const float*)d_in[12];
  float* out = (float*)d_out;     // [B*G*3 centers][B*G*128 features]

  char* ws = (char*)d_ws;
  size_t off = 0;
  auto take = [&](size_t bytes) -> void* {
    void* p = ws + off;
    off += (bytes + 255) & ~(size_t)255;
    return p;
  };
  _Float16* f1h    = (_Float16*)take((size_t)MG * KNB * C1_ * 2);   //  64 MB
  _Float16* f3h    = (_Float16*)take((size_t)MG * KNB * C3_ * 2);   // 256 MB
  _Float16* W2h    = (_Float16*)take((size_t)C2_ * C1_ * 2);
  _Float16* W3h    = (_Float16*)take((size_t)C3_ * C3_ * 2);
  _Float16* W4h    = (_Float16*)take((size_t)C4_ * C3_ * 2);
  int*      cidx   = (int*)take((size_t)BB * GG * 4);
  int*      knnidx = (int*)take((size_t)BB * GG * KNB * 4);
  float*    p1S    = (float*)take((size_t)MG * C1_ * 4);            //   4 MB
  float*    p1Q    = (float*)take((size_t)MG * C1_ * 4);
  float*    p2S    = (float*)take((size_t)MG * C3_ * 4);            //  16 MB
  float*    p2Q    = (float*)take((size_t)MG * C3_ * 4);
  float*    m1S    = (float*)take(32 * C1_ * 4);
  float*    m1Q    = (float*)take(32 * C1_ * 4);
  float*    m2S    = (float*)take(32 * C3_ * 4);
  float*    m2Q    = (float*)take(32 * C3_ * 4);
  float*    s1     = (float*)take(C1_ * 4);
  float*    h1     = (float*)take(C1_ * 4);
  float*    s2     = (float*)take(C3_ * 4);
  float*    h2     = (float*)take(C3_ * 4);

  const float invCnt = 1.0f / ((float)MG * (float)KNB);

  // Weight down-conversion (once per call; tiny).
  f32_to_f16_kernel<<<(C2_ * C1_ + 255) / 256, 256, 0, stream>>>(W2, W2h, C2_ * C1_);
  f32_to_f16_kernel<<<(C3_ * C3_ + 255) / 256, 256, 0, stream>>>(W3, W3h, C3_ * C3_);
  f32_to_f16_kernel<<<(C4_ * C3_ + 255) / 256, 256, 0, stream>>>(W4, W4h, C4_ * C3_);

  // Grouping.
  fps_kernel<<<BB, 256, 0, stream>>>(xyz, cidx, out);
  knn_kernel<<<BB * GG, 256, 0, stream>>>(xyz, out /*centers*/, knnidx);

  // Encoder.
  gather_l1_kernel<<<MG, 128, 0, stream>>>(xyz, knnidx, W1, b1, f1h, p1S, p1Q);
  bn_reduce_kernel<<<(C1_ * 32 + 255) / 256, 256, 0, stream>>>(p1S, p1Q, m1S, m1Q, C1_, MG / 32);
  bn_finalize_kernel<<<1, C1_, 0, stream>>>(m1S, m1Q, g1, be1, s1, h1, C1_, invCnt);

  encoder_mid_kernel<<<MG, 256, 0, stream>>>(f1h, W2h, b2, W3h, b3, s1, h1, f3h, p2S, p2Q);
  bn_reduce_kernel<<<(C3_ * 32 + 255) / 256, 256, 0, stream>>>(p2S, p2Q, m2S, m2Q, C3_, MG / 32);
  bn_finalize_kernel<<<2, 256, 0, stream>>>(m2S, m2Q, g2, be2, s2, h2, C3_, invCnt);

  encoder_out_kernel<<<MG, 256, 0, stream>>>(f3h, W4h, b4, s2, h2, out);
}
